// CIM_layer_50500225466529
// MI455X (gfx1250) — matrix operand order repaired and checked by reading, not compiled, not run on hardware
//
#include <hip/hip_runtime.h>

#define N     6144
#define C     20
#define K_TOP 615           // ceil(0.1 * 6144)
#define KPAD  640
#define IOU_THR 0.25f

typedef __attribute__((ext_vector_type(16))) _Float16 v16h;
typedef __attribute__((ext_vector_type(8)))  float    v8f;

// ---------------------------------------------------------------------------
// 1) preds[n][c] = predict_cls[n][c+1] * predict_det[n][c+1]
// ---------------------------------------------------------------------------
__global__ void k_preds(const float* __restrict__ cls, const float* __restrict__ det,
                        float* __restrict__ preds) {
    int id = blockIdx.x * blockDim.x + threadIdx.x;
    if (id >= N * C) return;
    int n = id / C, c = id % C;
    preds[id] = cls[n * (C + 1) + c + 1] * det[n * (C + 1) + c + 1];
}

// ---------------------------------------------------------------------------
// 2) top-K by rank counting. Stable order: (value desc, index asc) => ranks
//    are a permutation, so order[rank] writes are conflict-free.
//    grid = (C, 8); each block handles 768 candidate rows, full column in LDS.
// ---------------------------------------------------------------------------
__global__ void __launch_bounds__(1024) k_topk(const float* __restrict__ preds,
                                               int* __restrict__ order) {
    __shared__ float sval[N];
    const int c = blockIdx.x;
    for (int i = threadIdx.x; i < N; i += 1024)
        sval[i] = preds[i * C + c];
    __syncthreads();
    const int i = blockIdx.y * 768 + threadIdx.x;
    if (threadIdx.x < 768) {
        const float v = sval[i];
        int rank = 0;
#pragma unroll 4
        for (int j = 0; j < N; ++j) {
            float vj = sval[j];
            rank += (vj > v) || (vj == v && j < i);
        }
        if (rank < K_TOP) order[c * KPAD + rank] = i;
    }
}

// ---------------------------------------------------------------------------
// 3) greedy NMS per class. One block per class; order/alive/keep live in LDS.
//    Sequential over 615 pivots; suppression gather only when pivot alive.
// ---------------------------------------------------------------------------
__global__ void __launch_bounds__(640) k_nms(const float* __restrict__ iou,
                                             const int* __restrict__ order,
                                             unsigned char* __restrict__ mask) {
    __shared__ int sOrder[K_TOP];
    __shared__ int sAlive[K_TOP];
    __shared__ int sKeep[K_TOP];
    __shared__ int sFlag;
    __shared__ int sRow;
    const int c = blockIdx.x;
    const int tid = threadIdx.x;
    for (int n = tid; n < N; n += 640) mask[(size_t)c * N + n] = 0;
    if (tid < K_TOP) {
        sOrder[tid] = order[c * KPAD + tid];
        sAlive[tid] = 1;
        sKeep[tid]  = 0;
    }
    __syncthreads();
    for (int i = 0; i < K_TOP; ++i) {
        if (tid == 0) {
            int a = sAlive[i];
            sFlag = a;
            sKeep[i] = a;
            sRow = sOrder[i];
        }
        __syncthreads();
        if (sFlag && tid < K_TOP) {
            float v = iou[(size_t)sRow * N + sOrder[tid]];
            if (v >= IOU_THR) sAlive[tid] = 0;   // diag==1.0 also clears pivot (ok)
        }
        __syncthreads();
    }
    if (tid < K_TOP && sKeep[tid]) mask[(size_t)c * N + sOrder[tid]] = 1;
}

// ---------------------------------------------------------------------------
// 4) per-row sequential (in c) gt update: upd = mask & (p > w) & (labels[c]>0)
// ---------------------------------------------------------------------------
__global__ void k_gt(const float* __restrict__ preds, const unsigned char* __restrict__ mask,
                     const int* __restrict__ labels, int* __restrict__ gtcls,
                     float* __restrict__ gtw, unsigned char* __restrict__ gmask) {
    int n = blockIdx.x * blockDim.x + threadIdx.x;
    if (n >= N) return;
    float w = -1.0f;
    int cl = 0;
#pragma unroll
    for (int c = 0; c < C; ++c) {
        float p = preds[n * C + c];
        if (mask[(size_t)c * N + n] && (p > w) && (labels[c] > 0)) { cl = c + 1; w = p; }
    }
    gtcls[n] = cl;
    gtw[n]   = w;
    gmask[n] = (cl > 0) ? 1 : 0;
}

// ---------------------------------------------------------------------------
// 5) masked row max + first-occurrence argmax over iou_map (the 151MB pass).
//    One block per row, float4 streaming, gmask cached in LDS.
// ---------------------------------------------------------------------------
__global__ void __launch_bounds__(256) k_rowmax(const float* __restrict__ iou,
                                                const unsigned char* __restrict__ gmask,
                                                float* __restrict__ rowv,
                                                int* __restrict__ rowi) {
    __shared__ int   sMask[N / 4];
    __shared__ float sV[256];
    __shared__ int   sI[256];
    const int n = blockIdx.x;
    const int tid = threadIdx.x;
    const int* gm4 = (const int*)gmask;
    for (int k = tid; k < N / 4; k += 256) sMask[k] = gm4[k];
    __syncthreads();
    const unsigned char* sm = (const unsigned char*)sMask;
    const float4* row = (const float4*)(iou + (size_t)n * N);
    float bv = -2.0f;           // strictly below -1
    int   bi = 0x7fffffff;
#pragma unroll
    for (int k = 0; k < (N / 4) / 256; ++k) {      // 6 iterations
        int q = tid + k * 256;
        if (k < (N / 4) / 256 - 1)
            __builtin_prefetch((const void*)(row + q + 256), 0, 0);
        float4 f = row[q];
        int j0 = q << 2;
        float e;
        e = sm[j0 + 0] ? f.x : -1.0f; if (e > bv) { bv = e; bi = j0 + 0; }
        e = sm[j0 + 1] ? f.y : -1.0f; if (e > bv) { bv = e; bi = j0 + 1; }
        e = sm[j0 + 2] ? f.z : -1.0f; if (e > bv) { bv = e; bi = j0 + 2; }
        e = sm[j0 + 3] ? f.w : -1.0f; if (e > bv) { bv = e; bi = j0 + 3; }
    }
    sV[tid] = bv; sI[tid] = bi;
    __syncthreads();
    for (int s = 128; s > 0; s >>= 1) {
        if (tid < s) {
            float ov = sV[tid + s]; int oi = sI[tid + s];
            if (ov > sV[tid] || (ov == sV[tid] && oi < sI[tid])) { sV[tid] = ov; sI[tid] = oi; }
        }
        __syncthreads();
    }
    if (tid == 0) { rowv[n] = sV[0]; rowi[n] = sI[0]; }
}

// ---------------------------------------------------------------------------
// 6) final outputs. One wave per 16-row tile. pseudo_labels materialized via
//    v_wmma_f32_16x16x32_f16: A = class-indicator one-hots (f16, exact),
//    B = identity selectors for cols 0..15 and 16..20; sel==31 => zero row.
// ---------------------------------------------------------------------------
__global__ void __launch_bounds__(32) k_out(const float* __restrict__ rowv,
                                            const int* __restrict__ rowi,
                                            const int* __restrict__ gtcls,
                                            const float* __restrict__ gtw,
                                            float* __restrict__ outLabels,
                                            float* __restrict__ outIou,
                                            float* __restrict__ outW) {
    __shared__ int sSel[16];
    const int n0 = blockIdx.x * 16;
    const int lane = threadIdx.x;
    if (lane < 16) {
        int n = n0 + lane;
        float mv = rowv[n];
        int   mi = rowi[n];
        int   gc = gtcls[mi];
        float gw = gtw[mi];
        bool ign = (mv == 0.0f);
        bool bg  = (mv < IOU_THR) && !ign;
        int sel;
        if (bg) sel = 0;
        else if (ign) sel = 31;
        else sel = (gc > 0) ? gc : 31;
        sSel[lane] = sel;
        outIou[n] = mv;
        outW[n]   = ign ? 0.0f : gw;
    }
    __syncthreads();
    const int m  = lane & 15;
    const int hi = lane >> 4;
    const int sc = sSel[m];
    v16h a, b1, b2;
#pragma unroll
    for (int h = 0; h < 16; ++h) {
        // A (16x32 f16): lane<16 halves -> K=0..7,16..23 ; lane>=16 -> K=8..15,24..31
        int ka = (h < 8) ? (hi * 8 + h) : (16 + hi * 8 + (h - 8));
        a[h] = (ka == sc) ? (_Float16)1.0f : (_Float16)0.0f;
        // B (32x16 f16): lane<16 halves -> K=0..7,16..23 ; lane>=16 -> K=8..15,24..31
        int kb = (h < 8) ? (hi * 8 + h) : (16 + hi * 8 + (h - 8));
        b1[h] = (kb == m)        ? (_Float16)1.0f : (_Float16)0.0f;   // select cols 0..15
        b2[h] = (kb == (m + 16)) ? (_Float16)1.0f : (_Float16)0.0f;   // select cols 16..31
    }
    v8f cz = {};
    v8f d1 = __builtin_amdgcn_wmma_f32_16x16x32_f16(false, a, false, b1, (short)0, cz, false, false);
    v8f d2 = __builtin_amdgcn_wmma_f32_16x16x32_f16(false, a, false, b2, (short)0, cz, false, false);
    const int col = m;
    const int rbase = n0 + hi * 8;
#pragma unroll
    for (int v = 0; v < 8; ++v) {
        int r = rbase + v;                      // D: VGPR v, lane -> M = v + 8*hi, col = lane&15
        outLabels[r * (C + 1) + col] = d1[v];
        if (col < 5) outLabels[r * (C + 1) + 16 + col] = d2[v];
    }
}

// ---------------------------------------------------------------------------
extern "C" void kernel_launch(void* const* d_in, const int* in_sizes, int n_in,
                              void* d_out, int out_size, void* d_ws, size_t ws_size,
                              hipStream_t stream) {
    (void)in_sizes; (void)n_in; (void)out_size; (void)ws_size;
    const float* predict_cls = (const float*)d_in[0];
    const float* predict_det = (const float*)d_in[1];
    // d_in[2] = rois (unused by the reference after the slice)
    const int*   labels      = (const int*)d_in[3];
    const float* iou         = (const float*)d_in[4];

    char* ws = (char*)d_ws;
    float*         preds = (float*)ws;         ws += sizeof(float) * N * C;
    int*           order = (int*)ws;           ws += sizeof(int) * C * KPAD;
    unsigned char* mask  = (unsigned char*)ws; ws += (size_t)C * N;          // multiple of 4
    int*           gtcls = (int*)ws;           ws += sizeof(int) * N;
    float*         gtw   = (float*)ws;         ws += sizeof(float) * N;
    unsigned char* gmask = (unsigned char*)ws; ws += N;                      // multiple of 4
    float*         rowv  = (float*)ws;         ws += sizeof(float) * N;
    int*           rowi  = (int*)ws;           ws += sizeof(int) * N;

    float* outLabels = (float*)d_out;                 // (N, C+1)
    float* outIou    = outLabels + (size_t)N * (C + 1);
    float* outW      = outIou + N;

    k_preds <<<(N * C + 255) / 256, 256, 0, stream>>>(predict_cls, predict_det, preds);
    k_topk  <<<dim3(C, 8), 1024, 0, stream>>>(preds, order);
    k_nms   <<<C, 640, 0, stream>>>(iou, order, mask);
    k_gt    <<<(N + 255) / 256, 256, 0, stream>>>(preds, mask, labels, gtcls, gtw, gmask);
    k_rowmax<<<N, 256, 0, stream>>>(iou, gmask, rowv, rowi);
    k_out   <<<N / 16, 32, 0, stream>>>(rowv, rowi, gtcls, gtw, outLabels, outIou, outW);
}
